// CausalSelfAttention_66992899883414
// MI455X (gfx1250) — matrix operand order repaired and checked by reading, not compiled, not run on hardware
//
#include <hip/hip_runtime.h>
#include <hip/hip_bf16.h>

// Problem constants (match reference)
#define TT 2048
#define NH 16
#define NKV 4
#define HD 128
#define KB 64          // keys staged per LDS block in flash attention

typedef __attribute__((ext_vector_type(16))) _Float16 v16h;
typedef __attribute__((ext_vector_type(8)))  _Float16 h8;
typedef __attribute__((ext_vector_type(4)))  _Float16 h4;
typedef __attribute__((ext_vector_type(8)))  float    v8f;
typedef __attribute__((ext_vector_type(4)))  float    f4;

__device__ __forceinline__ v8f wmma16(v16h a, v16h b, v8f c) {
  // D = A(16x32 f16) * B(32x16 f16) + C(16x16 f32)
  return __builtin_amdgcn_wmma_f32_16x16x32_f16(false, a, false, b, (short)0, c,
                                                false, false);
}

__device__ __forceinline__ v8f vzero8() {
  v8f z = {0.f, 0.f, 0.f, 0.f, 0.f, 0.f, 0.f, 0.f};
  return z;
}

union U16 { v16h v; h8 h[2]; };

// A fragment (16x32, row-major source with leading dim ldk halves).
// Lane layout per ISA 7.12.2: lanes 0-15 row=lane, K = ks+{0..7, 16..23};
// lanes 16-31 row=lane-16, K = ks+{8..15, 24..31}.
__device__ __forceinline__ v16h frag_a(const _Float16* base, int ldk, int row0, int ks) {
  const int lane = threadIdx.x & 31;
  const int r = row0 + (lane & 15);
  const int kh = ks + ((lane >> 4) << 3);
  U16 u;
  u.h[0] = *(const h8*)(base + r * ldk + kh);
  u.h[1] = *(const h8*)(base + r * ldk + kh + 16);
  return u.v;
}

// B fragment (32x16) from a column-major ("Bt") tile: baseT[col*ldk + k].
// lanes 0-15: col=lane, K = ks+{0..7, 16..23} ; lanes 16-31: col=lane-16, K = ks+{8..15, 24..31}.
__device__ __forceinline__ v16h frag_b(const _Float16* baseT, int ldk, int col0, int ks) {
  const int lane = threadIdx.x & 31;
  const int c = col0 + (lane & 15);
  const int kh = ks + ((lane >> 4) << 3);
  U16 u;
  u.h[0] = *(const h8*)(baseT + c * ldk + kh);
  u.h[1] = *(const h8*)(baseT + c * ldk + kh + 16);
  return u.v;
}

// Async L2->LDS copy of 16 bytes per lane (CDNA5 GLOBAL_LOAD_ASYNC_TO_LDS,
// tracked by ASYNCcnt). lds_off is the wave-relative LDS byte address.
__device__ __forceinline__ void async_g2l_b128(unsigned lds_off, const void* gaddr) {
  asm volatile("global_load_async_to_lds_b128 %0, %1, off"
               :: "v"(lds_off), "v"(gaddr) : "memory");
}

template <int N>
__device__ __forceinline__ void wait_async() {
  asm volatile("s_wait_asynccnt %0" :: "n"(N) : "memory");
}

// Shared dynamic LDS symbol (used by gemm_f16 and flash_attn)
extern __shared__ _Float16 smem[];

// ---------------------------------------------------------------------------
// One-time converts: x -> f16, W (KxN f32) -> Wt (NxK f16, transposed)
// ---------------------------------------------------------------------------
__global__ void cvt_f32_f16(const float* __restrict__ src,
                            _Float16* __restrict__ dst) {
  size_t i = ((size_t)blockIdx.x * blockDim.x + threadIdx.x) << 2;
  f4 v = *(const f4*)(src + i);
  h4 h = {(_Float16)v[0], (_Float16)v[1], (_Float16)v[2], (_Float16)v[3]};
  *(h4*)(dst + i) = h;
}

__global__ __launch_bounds__(256) void transpose_w(const float* __restrict__ W,
                                                   _Float16* __restrict__ Wt,
                                                   int K, int N) {
  __shared__ _Float16 Ws[64 * 72];   // [k][n], padded stride
  const int tid = threadIdx.x;
  const int k0 = blockIdx.y << 6, n0 = blockIdx.x << 6;
  // Read 64x64 f32 tile, rows coalesced, convert to f16 into LDS
#pragma unroll
  for (int c = 0; c < 4; ++c) {
    int idx = tid + (c << 8);                 // 0..1023 float4 chunks
    int r = idx >> 4, c4 = (idx & 15) << 2;
    f4 v = *(const f4*)(W + (size_t)(k0 + r) * N + n0 + c4);
#pragma unroll
    for (int i = 0; i < 4; ++i) Ws[r * 72 + c4 + i] = (_Float16)v[i];
  }
  __syncthreads();
  // Write transposed: Wt[n][k], 16B-vectorized along k
#pragma unroll
  for (int c = 0; c < 2; ++c) {
    int idx = tid + (c << 8);                 // 0..511 h8 chunks
    int nr = idx >> 3, kc = (idx & 7) << 3;
    h8 o;
#pragma unroll
    for (int e = 0; e < 8; ++e) o[e] = Ws[(kc + e) * 72 + nr];
    *(h8*)(Wt + (size_t)(n0 + nr) * K + k0 + kc) = o;
  }
}

// ---------------------------------------------------------------------------
// WMMA GEMM on f16 operands: C(MxN) = A(MxK,f16) * Bt(NxK,f16)^T
// 256 threads / 8 waves, 128x128 C tile, BK=64. Tiles staged with
// GLOBAL_LOAD_ASYNC_TO_LDS_B128, double-buffered on ASYNCcnt.
// LDS (halves): buf0 A @0, buf0 B @8192, buf1 A @16384, buf1 B @24576 (64 KB)
// ---------------------------------------------------------------------------
__device__ __forceinline__ void stage_tiles(const _Float16* A, const _Float16* Bt,
                                            int K, int m0, int n0, int kk,
                                            unsigned bufbase /*halves*/) {
  const int tid = threadIdx.x;
#pragma unroll
  for (int c = 0; c < 4; ++c) {
    int idx = tid + (c << 8);                 // 0..1023 h8 chunks (A: 128x64)
    int mm = idx >> 3, k8 = (idx & 7) << 3;
    async_g2l_b128((bufbase + mm * 64 + k8) * 2,
                   A + (size_t)(m0 + mm) * K + kk + k8);
  }
#pragma unroll
  for (int c = 0; c < 4; ++c) {
    int idx = tid + (c << 8);                 // B tile: 128 n-rows x 64 k
    int nn = idx >> 3, k8 = (idx & 7) << 3;
    async_g2l_b128((bufbase + 8192 + nn * 64 + k8) * 2,
                   Bt + (size_t)(n0 + nn) * K + kk + k8);
  }
}

template <typename CT>
__global__ __launch_bounds__(256) void gemm_f16(const _Float16* __restrict__ A,
                                                const _Float16* __restrict__ Bt,
                                                CT* __restrict__ C,
                                                int M, int N, int K) {
  const int tid = threadIdx.x;
  const int lane = tid & 31, w = tid >> 5;
  const int m0 = blockIdx.y << 7, n0 = blockIdx.x << 7;

  v8f acc[8];
#pragma unroll
  for (int j = 0; j < 8; ++j) acc[j] = vzero8();

  stage_tiles(A, Bt, K, m0, n0, 0, 0u);       // prologue: buf0
  for (int kk = 0; kk < K; kk += 64) {
    const unsigned cur = ((kk >> 6) & 1) ? 16384u : 0u;
    const bool nxt = (kk + 64 < K);
    if (nxt) stage_tiles(A, Bt, K, m0, n0, kk + 64, cur ^ 16384u);
    if (nxt) wait_async<8>(); else wait_async<0>();
    __syncthreads();
    const _Float16* As = smem + cur;
    const _Float16* Bs = smem + cur + 8192;
#pragma unroll
    for (int ks = 0; ks < 64; ks += 32) {
      v16h a = frag_a(As, 64, w << 4, ks);
#pragma unroll
      for (int j = 0; j < 8; ++j)
        acc[j] = wmma16(a, frag_b(Bs, 64, j << 4, ks), acc[j]);
    }
    __syncthreads();                          // readers done before re-stage
  }
  const int g = lane >> 4, cl = lane & 15;
#pragma unroll
  for (int j = 0; j < 8; ++j)
#pragma unroll
    for (int i = 0; i < 8; ++i)
      C[(size_t)(m0 + (w << 4) + (g << 3) + i) * N + n0 + (j << 4) + cl] =
          (CT)acc[j][i];
}

// ---------------------------------------------------------------------------
// RoPE (f16 in / f16 out) + (B,T,H,D)->(B,H,T,D) transpose.
// One thread per (b,t,h, pair d<64): handles elements d and d+64.
// ---------------------------------------------------------------------------
__global__ void rope_q_kernel(const _Float16* __restrict__ Qb,
                              const int* __restrict__ pos,
                              _Float16* __restrict__ Qh) {
  int idx = blockIdx.x * blockDim.x + threadIdx.x;  // B*T*16*64 = 2^23
  int d = idx & 63;
  int h = (idx >> 6) & 15;
  int t = (idx >> 10) & (TT - 1);
  int b = idx >> 21;
  float invf = __expf(-(float)d * (1.0f / 64.0f) * 9.210340371976184f); // ln(1e4)
  float ang = (float)pos[t] * invf;
  float c = __cosf(ang), s = __sinf(ang);
  const _Float16* src = Qb + (((size_t)(b * TT + t) * NH + h) << 7);
  float q0 = (float)src[d], q1 = (float)src[d + 64];
  _Float16* dst = Qh + (((size_t)(b * NH + h) * TT + t) << 7);
  dst[d]      = (_Float16)(q0 * c - q1 * s);
  dst[d + 64] = (_Float16)(q1 * c + q0 * s);
}

__global__ void rope_kv_kernel(const _Float16* __restrict__ Kb,
                               const _Float16* __restrict__ Vb,
                               const int* __restrict__ pos,
                               _Float16* __restrict__ Kh,
                               _Float16* __restrict__ Vh) {
  int idx = blockIdx.x * blockDim.x + threadIdx.x;  // B*T*4*64 = 2^21
  int d = idx & 63;
  int h = (idx >> 6) & 3;
  int t = (idx >> 8) & (TT - 1);
  int b = idx >> 19;
  float invf = __expf(-(float)d * (1.0f / 64.0f) * 9.210340371976184f);
  float ang = (float)pos[t] * invf;
  float c = __cosf(ang), s = __sinf(ang);
  size_t srcoff = ((size_t)(b * TT + t) * NKV + h) << 7;
  size_t dstoff = ((size_t)(b * NKV + h) * TT + t) << 7;
  float k0 = (float)Kb[srcoff + d], k1 = (float)Kb[srcoff + d + 64];
  Kh[dstoff + d]      = (_Float16)(k0 * c - k1 * s);
  Kh[dstoff + d + 64] = (_Float16)(k1 * c + k0 * s);
  Vh[dstoff + d]      = Vb[srcoff + d];
  Vh[dstoff + d + 64] = Vb[srcoff + d + 64];
}

// ---------------------------------------------------------------------------
// Flash attention: one block per (b, h, 128 q rows); 8 waves x 16 rows each.
// Dynamic LDS only (LDS byte offsets are plain arithmetic for async ops):
//   Ks = [key][d]   (64x128 f16, 16 KB) -- GLOBAL_LOAD_ASYNC_TO_LDS
//   Vt = [d][key]   (128x64 f16, 16 KB) -- VGPR-mediated transpose staging
//   Ps = per-wave P scratch (8x16x32 f16, 8 KB)
// Online softmax in f32; f16 output feeds the Wo WMMA GEMM directly.
// ---------------------------------------------------------------------------
#define KS_OFF 0
#define VT_OFF (KB * 128)
#define PS_OFF (2 * KB * 128)

__global__ __launch_bounds__(256) void flash_attn(const _Float16* __restrict__ Qh,
                                                  const _Float16* __restrict__ Kh,
                                                  const _Float16* __restrict__ Vh,
                                                  _Float16* __restrict__ O) {
  _Float16* Ks = smem + KS_OFF;
  _Float16* Vt = smem + VT_OFF;
  _Float16* Ps = smem + PS_OFF;
  const int tid = threadIdx.x, lane = tid & 31, w = tid >> 5;
  const int g = lane >> 4, cl = lane & 15;
  const int qblk = blockIdx.x, h = blockIdx.y, b = blockIdx.z;
  const int kvh = h >> 2;                       // repeat_interleave: h -> h/4
  const int qrow0 = (qblk << 7) + (w << 4);     // wave's first q row (abs t)

  // Q fragments for this wave: 16 rows x 128 d, kept in registers
  const _Float16* qbase = Qh + (((size_t)(b * NH + h) * TT + (qblk << 7)) << 7);
  v16h qf[4];
#pragma unroll
  for (int kc = 0; kc < 4; ++kc) qf[kc] = frag_a(qbase, 128, w << 4, kc << 5);

  v8f o[8];
#pragma unroll
  for (int j = 0; j < 8; ++j) o[j] = vzero8();
  float m[8], l[8];
#pragma unroll
  for (int i = 0; i < 8; ++i) { m[i] = -1e30f; l[i] = 0.f; }

  const _Float16* kbase = Kh + (((size_t)(b * NKV + kvh) * TT) << 7);
  const _Float16* vbase = Vh + (((size_t)(b * NKV + kvh) * TT) << 7);
  const int nkb = (qblk << 1) + 2;  // 64-key blocks covering [0,(qblk+1)*128)

  for (int kb = 0; kb < nkb; ++kb) {
    // Stage K tile (64 keys x 128 d) with async global->LDS copies (16B/lane)
#pragma unroll
    for (int c = 0; c < 4; ++c) {
      int idx = tid + (c << 8);               // 0..1023
      int key = idx >> 4, d8 = (idx & 15) << 3;
      unsigned loff = (unsigned)((KS_OFF + key * 128 + d8) * 2);
      async_g2l_b128(loff, kbase + ((size_t)(kb * KB + key) << 7) + d8);
    }
    // Stage V tile transposed into Vt[d][key] (overlaps with async K copies)
#pragma unroll
    for (int c = 0; c < 4; ++c) {
      int idx = tid + (c << 8);
      int key = idx >> 4, d8 = (idx & 15) << 3;
      h8 v = *(const h8*)(vbase + ((size_t)(kb * KB + key) << 7) + d8);
#pragma unroll
      for (int i = 0; i < 8; ++i) Vt[(d8 + i) * KB + key] = v[i];
    }
    wait_async<0>();
    __syncthreads();

    // Warm L2 for the next tile while this one computes
    if (kb + 1 < nkb) {
      __builtin_prefetch(kbase + ((size_t)((kb + 1) * KB) << 7), 0, 1);
      __builtin_prefetch(vbase + ((size_t)((kb + 1) * KB) << 7), 0, 1);
    }

    for (int kt = 0; kt < KB; kt += 32) {
      const int keybase = kb * KB + kt;
      if (keybase <= qrow0 + 15) {            // per-wave causal skip (uniform)
        // Scores: S(16x32) = Q(16x128) * K^T, two 16-key halves
        v8f s0 = vzero8(), s1 = vzero8();
#pragma unroll
        for (int kc = 0; kc < 4; ++kc) {
          s0 = wmma16(qf[kc], frag_b(Ks, 128, kt, kc << 5), s0);
          s1 = wmma16(qf[kc], frag_b(Ks, 128, kt + 16, kc << 5), s1);
        }
        // Online softmax (rows: lanes 0-15 -> 0-7, lanes 16-31 -> 8-15)
        float corr[8];
#pragma unroll
        for (int i = 0; i < 8; ++i) {
          const int qi = qrow0 + (g << 3) + i;
          const int k0 = keybase + cl;
          float a0 = (k0      <= qi) ? s0[i] * 0.08838834764831845f : -1e30f;
          float a1 = (k0 + 16 <= qi) ? s1[i] * 0.08838834764831845f : -1e30f;
          float t = fmaxf(a0, a1);
          t = fmaxf(t, __shfl_xor(t, 1));
          t = fmaxf(t, __shfl_xor(t, 2));
          t = fmaxf(t, __shfl_xor(t, 4));
          t = fmaxf(t, __shfl_xor(t, 8));
          float mn = fmaxf(m[i], t);
          float cr = __expf(m[i] - mn);
          m[i] = mn;
          float p0 = __expf(a0 - mn);
          float p1 = __expf(a1 - mn);
          float rs = p0 + p1;
          rs += __shfl_xor(rs, 1);
          rs += __shfl_xor(rs, 2);
          rs += __shfl_xor(rs, 4);
          rs += __shfl_xor(rs, 8);
          l[i] = l[i] * cr + rs;
          corr[i] = cr;
          _Float16* pp = Ps + (w << 9) + (((g << 3) + i) << 5) + cl;
          pp[0]  = (_Float16)p0;
          pp[16] = (_Float16)p1;
        }
#pragma unroll
        for (int j = 0; j < 8; ++j)
#pragma unroll
          for (int i = 0; i < 8; ++i) o[j][i] *= corr[i];
        // P(16x32) * V(32x128): A-frag from per-wave LDS scratch
        v16h pf = frag_a(Ps + (w << 9), 32, 0, 0);
#pragma unroll
        for (int j = 0; j < 8; ++j)
          o[j] = wmma16(pf, frag_b(Vt, KB, j << 4, kt), o[j]);
      }
    }
    __syncthreads();
  }

  float invl[8];
#pragma unroll
  for (int i = 0; i < 8; ++i) invl[i] = 1.0f / l[i];
#pragma unroll
  for (int j = 0; j < 8; ++j)
#pragma unroll
    for (int i = 0; i < 8; ++i) {
      int t = qrow0 + (g << 3) + i;
      O[(size_t)(b * TT + t) * (NH * HD) + (h << 7) + (j << 4) + cl] =
          (_Float16)(o[j][i] * invl[i]);
    }
}

// ---------------------------------------------------------------------------
extern "C" void kernel_launch(void* const* d_in, const int* in_sizes, int n_in,
                              void* d_out, int out_size, void* d_ws, size_t ws_size,
                              hipStream_t stream) {
  (void)in_sizes; (void)n_in; (void)out_size; (void)ws_size;
  const float* x  = (const float*)d_in[0];
  const int*   pos = (const int*)d_in[1];
  const float* Wq = (const float*)d_in[2];
  const float* Wk = (const float*)d_in[3];
  const float* Wv = (const float*)d_in[4];
  const float* Wo = (const float*)d_in[5];
  float* out = (float*)d_out;

  const size_t MR = (size_t)4 * TT;  // 8192 rows
  char* p = (char*)d_ws;
  _Float16* xh  = (_Float16*)p; p += MR * 2048 * 2;       // 32 MB
  _Float16* Wqt = (_Float16*)p; p += (size_t)2048 * 2048 * 2;  // 8 MB
  _Float16* Wkt = (_Float16*)p; p += (size_t)512 * 2048 * 2;   // 2 MB
  _Float16* Wvt = (_Float16*)p; p += (size_t)512 * 2048 * 2;   // 2 MB
  _Float16* Wot = (_Float16*)p; p += (size_t)2048 * 2048 * 2;  // 8 MB
  _Float16* Qb  = (_Float16*)p; p += MR * 2048 * 2;       // 32 MB (b,t,h,d)
  _Float16* Kb  = (_Float16*)p; p += MR * 512 * 2;        // 8 MB
  _Float16* Vb  = (_Float16*)p; p += MR * 512 * 2;        // 8 MB
  _Float16* Qh  = (_Float16*)p; p += MR * 2048 * 2;       // 32 MB (b,h,t,d)
  _Float16* Kh  = (_Float16*)p; p += MR * 512 * 2;        // 8 MB
  _Float16* Vh  = (_Float16*)p; p += MR * 512 * 2;        // 8 MB
  _Float16* Oh  = (_Float16*)p; p += MR * 2048 * 2;       // 32 MB

  // One-time f16 conversion / weight transposes
  cvt_f32_f16<<<(MR * 2048 / 4) / 256, 256, 0, stream>>>(x, xh);
  transpose_w<<<dim3(32, 32), 256, 0, stream>>>(Wq, Wqt, 2048, 2048);
  transpose_w<<<dim3(8, 32), 256, 0, stream>>>(Wk, Wkt, 2048, 512);
  transpose_w<<<dim3(8, 32), 256, 0, stream>>>(Wv, Wvt, 2048, 512);
  transpose_w<<<dim3(32, 32), 256, 0, stream>>>(Wo, Wot, 2048, 2048);

  // QKV projections (async double-buffered WMMA GEMM, 64 KB dynamic LDS)
  gemm_f16<_Float16><<<dim3(16, 64), 256, 65536, stream>>>(xh, Wqt, Qb, 8192, 2048, 2048);
  gemm_f16<_Float16><<<dim3(4, 64), 256, 65536, stream>>>(xh, Wkt, Kb, 8192, 512, 2048);
  gemm_f16<_Float16><<<dim3(4, 64), 256, 65536, stream>>>(xh, Wvt, Vb, 8192, 512, 2048);
  // RoPE + head-major transpose
  rope_q_kernel<<<(4 * TT * NH * 64) / 256, 256, 0, stream>>>(Qb, pos, Qh);
  rope_kv_kernel<<<(4 * TT * NKV * 64) / 256, 256, 0, stream>>>(Kb, Vb, pos, Kh, Vh);
  // Flash attention (dynamic LDS: Ks + Vt + Ps = 40960 bytes), f16 out
  flash_attn<<<dim3(TT / 128, NH, 4), 256, 40960, stream>>>(Qh, Kh, Vh, Oh);
  // Output projection -> f32 result
  gemm_f16<float><<<dim3(16, 64), 256, 65536, stream>>>(Oh, Wot, out, 8192, 2048, 2048);
}
